// RA_42855183679635
// MI455X (gfx1250) — matrix operand order repaired and checked by reading, not compiled, not run on hardware
//
#include <hip/hip_runtime.h>
#include <hip/hip_bf16.h>

// ---------------- problem constants ----------------
#define L     128
#define EMB   300
#define T     256
#define S     256
#define KTOP  10
#define N     8192
#define G3    768          // 3*T = 3*S
// output layout: sc(1), h_new(256), vs_new(8193*256), hs_new(8193*256)
#define OFF_H   1
#define OFF_VS  257
#define OFF_HS  (257 + 8193*256)

typedef __attribute__((ext_vector_type(16))) _Float16 v16h;
typedef __attribute__((ext_vector_type(4)))  _Float16 v4h;
typedef __attribute__((ext_vector_type(8)))  float    v8f;
typedef __attribute__((ext_vector_type(4)))  unsigned int u32x4;

union V16 { u32x4 q[2]; v16h h; };

// ============================================================
// WMMA projection:  out[m][n] = bias[n] + sum_k A[m][k] * W[n][k]
//   MODE 0: A[m][k] = emb[topic[m]][k]        (KACT=300, pad to 320)
//   MODE 1: A[m][k] = emb[topic[127-m]][k]    (KACT=300, pad to 320)
//   MODE 2: A[m][k] = k<256 ? yf[m][k] : yb[m][k-256]   (KACT=512)
//
// block = 128 threads (4 waves), tile = 16(M) x 64(N).
// Software-pipelined, double-buffered LDS: chunk kc+1's global_load_b128s
// are issued between the barrier and the WMMA of chunk kc, hiding memory
// latency under ds_load+wmma; one barrier per chunk.  Fully unrolled so
// all mode/tail selection folds to constants.
// ============================================================

template<int MODE, int KACT>
__device__ __forceinline__ void issue_loads(int kbase,
                                            const float* a_src, int a_mrow, int a_kk,
                                            const float* yf, const float* yb,
                                            const float* b_src, int b_seg,
                                            float4& pa, float4 (&pb)[4])
{
    if (kbase + 32 <= KACT) {                   // folds: full chunks
        const float* ap;
        if (MODE == 2) {
            // kbase is a constant per unrolled iteration: whole chunk is
            // either in yf (kbase<256) or yb (kbase>=256); never mixed.
            const float* src = (kbase < T) ? yf : yb;
            ap = src + (size_t)a_mrow * T + (kbase - ((kbase < T) ? 0 : T)) + a_kk;
        } else {
            ap = a_src + kbase + a_kk;
        }
        pa = *(const float4*)ap;
#pragma unroll
        for (int i = 0; i < 4; ++i)
            pb[i] = *(const float4*)(b_src + kbase + b_seg + 4*i);
    } else {                                    // K tail (emb modes only)
        float t0[4];
#pragma unroll
        for (int j = 0; j < 4; ++j) {
            const int k = kbase + a_kk + j;
            t0[j] = (k < KACT) ? a_src[k] : 0.0f;
        }
        pa = make_float4(t0[0], t0[1], t0[2], t0[3]);
#pragma unroll
        for (int i = 0; i < 4; ++i) {
            float t1[4];
#pragma unroll
            for (int j = 0; j < 4; ++j) {
                const int k = kbase + b_seg + 4*i + j;
                t1[j] = (k < KACT) ? b_src[k] : 0.0f;
            }
            pb[i] = make_float4(t1[0], t1[1], t1[2], t1[3]);
        }
    }
}

__device__ __forceinline__ v4h cvt4(float4 f)
{
    v4h h;
    h.x = (_Float16)f.x; h.y = (_Float16)f.y;
    h.z = (_Float16)f.z; h.w = (_Float16)f.w;
    return h;
}

template<int MODE, int KACT, int NKC>
__global__ __launch_bounds__(128)
void wmma_proj_kernel(const float* __restrict__ emb,
                      const int*   __restrict__ topic,
                      const float* __restrict__ yf,
                      const float* __restrict__ yb,
                      const float* __restrict__ W,
                      const float* __restrict__ bias,
                      float* __restrict__ out)
{
    __shared__ _Float16 Ah[2][16][40];          // double-buffered tiles
    __shared__ _Float16 Bh[2][4][16][40];

    const int tid   = threadIdx.x;              // 0..127
    const int lane  = tid & 31;
    const int wave  = tid >> 5;                 // 0..3
    const int nblk  = blockIdx.x;               // 0..11  (768 / 64)
    const int mtile = blockIdx.y;               // 0..7   (128 / 16)

    // ---- staging assignment: A = 512 floats / 128 threads = 4 each ----
    const int a_idx  = tid * 4;
    const int a_row  = a_idx >> 5;              // 0..15
    const int a_kk   = a_idx & 31;              // 0,4,...,28
    const int a_mrow = mtile * 16 + a_row;
    const float* a_src = nullptr;
    if (MODE == 0)      a_src = emb + (size_t)topic[a_mrow]       * EMB;
    else if (MODE == 1) a_src = emb + (size_t)topic[(L-1)-a_mrow] * EMB;

    // ---- staging assignment: B = per-wave 16 cols x 32 k, 16 f32/lane ----
    const int b_col  = lane >> 1;               // 0..15
    const int b_seg  = (lane & 1) * 16;         // 0 or 16
    const int b_ncol = nblk * 64 + wave * 16 + b_col;
    const float* b_src = W + (size_t)b_ncol * KACT;

    // ---- compute-phase LDS read offsets (ISA A/B layouts) ----
    const int mloc = lane & 15;
    const int aofs = (lane < 16) ? 0 : 8;       // A: +[0..7] and +[16..23]
    const int bofs = (lane < 16) ? 0 : 8;       // B: +[0..7] and +[16..23]

    v8f c = {};
    float4 pa; float4 pb[4];
    issue_loads<MODE, KACT>(0, a_src, a_mrow, a_kk, yf, yb, b_src, b_seg, pa, pb);

#pragma unroll
    for (int kc = 0; kc < NKC; ++kc) {
        const int buf = kc & 1;

        // ---- convert + store chunk kc into LDS buffer ----
        *(v4h*)&Ah[buf][a_row][a_kk] = cvt4(pa);
#pragma unroll
        for (int i = 0; i < 4; ++i)
            *(v4h*)&Bh[buf][wave][b_col][b_seg + 4*i] = cvt4(pb[i]);

        __syncthreads();

        // ---- prefetch chunk kc+1 (hides under ds_load + wmma) ----
        if (kc + 1 < NKC) {
            float4 na; float4 nb[4];
            issue_loads<MODE, KACT>((kc + 1) * 32, a_src, a_mrow, a_kk,
                                    yf, yb, b_src, b_seg, na, nb);
            pa = na;
#pragma unroll
            for (int i = 0; i < 4; ++i) pb[i] = nb[i];
        }

        // ---- consume chunk kc ----
        V16 ua, ub;
        ua.q[0] = *(const u32x4*)&Ah[buf][mloc][aofs];
        ua.q[1] = *(const u32x4*)&Ah[buf][mloc][aofs + 16];
        ub.q[0] = *(const u32x4*)&Bh[buf][wave][mloc][bofs];
        ub.q[1] = *(const u32x4*)&Bh[buf][wave][mloc][bofs + 16];
        c = __builtin_amdgcn_wmma_f32_16x16x32_f16(false, ua.h, false, ub.h,
                                                   (short)0, c, false, false);
        // note: the barrier at the top of iteration kc+1 orders this
        // iteration's LDS reads before buffer `buf` is overwritten at kc+2.
    }

    // ---------------- store D + bias ----------------
    const int ncol = nblk * 64 + wave * 16 + (lane & 15);
    const float bn = bias[ncol];
#pragma unroll
    for (int r = 0; r < 8; ++r) {
        const int m = mtile * 16 + r + ((lane < 16) ? 0 : 8);
        out[(size_t)m * G3 + ncol] = c[r] + bn;
    }
}

// ============================================================
// Sequential GRU scan, single workgroup of 768 threads.
//  G[t][768] = precomputed x@Wih^T + bih   (gates r,z,n)
//  store_mode 0: out[t][256]=h_t   1: out[127-t][256]=h_t   2: out[256]=h_last
// ============================================================
__global__ __launch_bounds__(G3)
void gru_scan_kernel(const float* __restrict__ G,
                     const float* __restrict__ whh,
                     const float* __restrict__ bhh,
                     float* __restrict__ out,
                     int store_mode)
{
    __shared__ float h[T];
    __shared__ float gh[G3];
    const int tid = threadIdx.x;            // 0..767
    if (tid < T) h[tid] = 0.0f;
    __syncthreads();

    const float4* wq = (const float4*)(whh + (size_t)tid * T);  // 16B-aligned
    const float  bh  = bhh[tid];

    for (int t = 0; t < L; ++t) {
        // prefetch next timestep's gate row (global_prefetch_b8)
        if (t + 1 < L)
            __builtin_prefetch(G + (size_t)(t + 1) * G3 + tid, 0, 3);

        float acc = bh;
        const float4* h4 = (const float4*)h;
#pragma unroll 4
        for (int k4 = 0; k4 < T / 4; ++k4) {
            const float4 w4 = wq[k4];       // global_load_b128
            const float4 hv = h4[k4];       // ds_load_b128
            acc += hv.x * w4.x + hv.y * w4.y + hv.z * w4.z + hv.w * w4.w;
        }
        gh[tid] = acc;
        __syncthreads();

        if (tid < T) {
            const float* g = G + (size_t)t * G3;
            const float r  = 1.0f / (1.0f + __expf(-(g[tid]       + gh[tid])));
            const float z  = 1.0f / (1.0f + __expf(-(g[T + tid]   + gh[T + tid])));
            const float n  = tanhf(g[2*T + tid] + r * gh[2*T + tid]);
            const float hn = (1.0f - z) * n + z * h[tid];
            h[tid] = hn;
            if (store_mode == 0)      out[(size_t)t         * T + tid] = hn;
            else if (store_mode == 1) out[(size_t)((L-1)-t) * T + tid] = hn;
        }
        __syncthreads();
    }
    if (store_mode == 2 && tid < T) out[tid] = h[tid];
}

// ============================================================
// alpha[row] = dot(vs[row], v) : one wave per row, coalesced.
// ============================================================
__global__ __launch_bounds__(256)
void alpha_kernel(const float* __restrict__ vs,
                  const float* __restrict__ v,
                  float* __restrict__ alpha)
{
    const int lane = threadIdx.x & 31;
    const int row  = blockIdx.x * (blockDim.x >> 5) + (threadIdx.x >> 5);
    if (row >= N) return;
    const float* r = vs + (size_t)row * T;
    float acc = 0.0f;
#pragma unroll
    for (int k = lane; k < T; k += 32) acc += r[k] * v[k];
#pragma unroll
    for (int off = 16; off > 0; off >>= 1)
        acc += __shfl_down(acc, off, 32);
    if (lane == 0) alpha[row] = acc;
}

// ============================================================
// finalize: top-K, softmax, attn, score head, GRU step, tail rows.
// single block of 256 threads.  alpha is a mutable workspace copy
// (rewritten every call by alpha_kernel, so destroying it is OK).
// ============================================================
__global__ __launch_bounds__(256)
void finalize_kernel(const float* __restrict__ v,
                     float* __restrict__ alpha,
                     const float* __restrict__ hs,
                     const float* __restrict__ h_old,
                     const float* __restrict__ score,
                     const float* __restrict__ w_ih_s,
                     const float* __restrict__ w_hh_s,
                     const float* __restrict__ b_ih_s,
                     const float* __restrict__ b_hh_s,
                     const float* __restrict__ score_w,
                     const float* __restrict__ score_b,
                     float* __restrict__ out)
{
    __shared__ float sv[256];
    __shared__ int   si[256];
    __shared__ float topw[KTOP];
    __shared__ int   topi[KTOP];
    __shared__ float xin[2*T + 1];
    const int tid = threadIdx.x;

    // ---- iterative top-K (argmax + invalidate), K=10 ----
    for (int kk = 0; kk < KTOP; ++kk) {
        float best = -3.4e38f; int bi = 0;
        for (int i = tid; i < N; i += 256) {
            const float a = alpha[i];
            if (a > best) { best = a; bi = i; }
        }
        sv[tid] = best; si[tid] = bi;
        __syncthreads();
        for (int s2 = 128; s2 > 0; s2 >>= 1) {
            if (tid < s2 && sv[tid + s2] > sv[tid]) {
                sv[tid] = sv[tid + s2]; si[tid] = si[tid + s2];
            }
            __syncthreads();
        }
        if (tid == 0) {
            topw[kk] = sv[0]; topi[kk] = si[0];
            alpha[si[0]] = -3.4e38f;
        }
        __syncthreads();
    }

    // ---- softmax over the K selected scores ----
    if (tid == 0) {
        float mx = topw[0];
        for (int k = 1; k < KTOP; ++k) mx = fmaxf(mx, topw[k]);
        float s = 0.0f;
        for (int k = 0; k < KTOP; ++k) { topw[k] = __expf(topw[k] - mx); s += topw[k]; }
        for (int k = 0; k < KTOP; ++k) topw[k] /= s;
    }
    __syncthreads();

    // ---- attn_h = sum_k w_k * hs[idx_k] ----
    float ah = 0.0f;
#pragma unroll
    for (int k = 0; k < KTOP; ++k) ah += topw[k] * hs[(size_t)topi[k] * S + tid];

    // ---- score head: pred_v = [v, attn_h, h_old, K] . score_w + score_b ----
    const float p = v[tid]     * score_w[tid]
                  + ah         * score_w[T + tid]
                  + h_old[tid] * score_w[2*T + tid];
    sv[tid] = p;
    __syncthreads();
    for (int s2 = 128; s2 > 0; s2 >>= 1) {
        if (tid < s2) sv[tid] += sv[tid + s2];
        __syncthreads();
    }
    if (tid == 0)
        out[0] = sv[0] + (float)KTOP * score_w[3*T] + score_b[0];

    // ---- GRU step, score_mode='double': x_in = [v*gp, v*gn, score] ----
    const float sc = score[0];
    const float gp = (sc >= 0.5f) ? 1.0f : 0.0f;
    xin[tid]       = v[tid] * gp;
    xin[T + tid]   = v[tid] * (1.0f - gp);
    if (tid == 0) xin[2*T] = sc;
    __syncthreads();

    float gi[3], gg[3];
#pragma unroll
    for (int g = 0; g < 3; ++g) {
        const int row = g * S + tid;
        float a1 = b_ih_s[row];
        const float* wr = w_ih_s + (size_t)row * (2*T + 1);
        for (int k = 0; k < 2*T + 1; ++k) a1 += xin[k] * wr[k];
        gi[g] = a1;
        float a2 = b_hh_s[row];
        const float* wr2 = w_hh_s + (size_t)row * S;
        for (int k = 0; k < S; ++k) a2 += h_old[k] * wr2[k];
        gg[g] = a2;
    }
    const float r  = 1.0f / (1.0f + __expf(-(gi[0] + gg[0])));
    const float z  = 1.0f / (1.0f + __expf(-(gi[1] + gg[1])));
    const float n  = tanhf(gi[2] + r * gg[2]);
    const float hn = (1.0f - z) * n + z * h_old[tid];

    out[OFF_H + tid] = hn;                                 // h_new
    out[OFF_HS + (size_t)N * S + tid] = hn;                // hs_new tail row
    out[OFF_VS + (size_t)N * T + tid] = v[tid];            // vs_new tail row
}

// ============================================================
extern "C" void kernel_launch(void* const* d_in, const int* in_sizes, int n_in,
                              void* d_out, int out_size, void* d_ws, size_t ws_size,
                              hipStream_t stream)
{
    const int*   topic   = (const int*)  d_in[0];
    const float* score   = (const float*)d_in[1];
    // d_in[2] = time (unused by reference)
    const float* h_old   = (const float*)d_in[3];
    const float* vs      = (const float*)d_in[4];
    const float* hs      = (const float*)d_in[5];
    const float* emb     = (const float*)d_in[6];
    const float* w_ih_f  = (const float*)d_in[7];
    const float* w_hh_f  = (const float*)d_in[8];
    const float* b_ih_f  = (const float*)d_in[9];
    const float* b_hh_f  = (const float*)d_in[10];
    const float* w_ih_b  = (const float*)d_in[11];
    const float* w_hh_b  = (const float*)d_in[12];
    const float* b_ih_b  = (const float*)d_in[13];
    const float* b_hh_b  = (const float*)d_in[14];
    const float* w_ih_o  = (const float*)d_in[15];
    const float* w_hh_o  = (const float*)d_in[16];
    const float* b_ih_o  = (const float*)d_in[17];
    const float* b_hh_o  = (const float*)d_in[18];
    const float* w_ih_s  = (const float*)d_in[19];
    const float* w_hh_s  = (const float*)d_in[20];
    const float* b_ih_s  = (const float*)d_in[21];
    const float* b_hh_s  = (const float*)d_in[22];
    const float* score_w = (const float*)d_in[23];
    const float* score_b = (const float*)d_in[24];
    float* out = (float*)d_out;

    // workspace layout (floats)
    float* Gf    = (float*)d_ws;            // 128*768
    float* Gb    = Gf + (size_t)L * G3;     // 128*768
    float* Go    = Gb + (size_t)L * G3;     // 128*768
    float* yf    = Go + (size_t)L * G3;     // 128*256
    float* yb    = yf + (size_t)L * T;      // 128*256
    float* v     = yb + (size_t)L * T;      // 256
    float* alpha = v + T;                   // 8192

    const dim3 wgrid(G3 / 64, L / 16);      // 12 x 8 blocks (4 waves each)
    const dim3 wblk(128);

    // 1) input projections for fwd/bwd word GRUs (WMMA, K=300 padded to 320)
    wmma_proj_kernel<0, EMB, 10><<<wgrid, wblk, 0, stream>>>(
        emb, topic, nullptr, nullptr, w_ih_f, b_ih_f, Gf);
    wmma_proj_kernel<1, EMB, 10><<<wgrid, wblk, 0, stream>>>(
        emb, topic, nullptr, nullptr, w_ih_b, b_ih_b, Gb);

    // 2) forward / backward scans (sequential, one persistent workgroup each)
    gru_scan_kernel<<<1, G3, 0, stream>>>(Gf, w_hh_f, b_hh_f, yf, 0);
    gru_scan_kernel<<<1, G3, 0, stream>>>(Gb, w_hh_b, b_hh_b, yb, 1);

    // 3) output-GRU input projection over y = [yf|yb]  (WMMA, K=512)
    wmma_proj_kernel<2, 2*T, 16><<<wgrid, wblk, 0, stream>>>(
        emb, topic, yf, yb, w_ih_o, b_ih_o, Go);

    // 4) output scan -> v (last hidden only)
    gru_scan_kernel<<<1, G3, 0, stream>>>(Go, w_hh_o, b_hh_o, v, 2);

    // 5) attention logits alpha = vs @ v   (wave per row, coalesced)
    alpha_kernel<<<N / 8, 256, 0, stream>>>(vs, v, alpha);

    // 6) history-cache copies (bulk of the bytes: 2 x 8 MB at HBM rate)
    hipMemcpyAsync(out + OFF_VS, vs, (size_t)N * T * sizeof(float),
                   hipMemcpyDeviceToDevice, stream);
    hipMemcpyAsync(out + OFF_HS, hs, (size_t)N * S * sizeof(float),
                   hipMemcpyDeviceToDevice, stream);

    // 7) top-K + softmax + attention + score + GRU step + tail rows
    finalize_kernel<<<1, 256, 0, stream>>>(v, alpha, hs, h_old, score,
                                           w_ih_s, w_hh_s, b_ih_s, b_hh_s,
                                           score_w, score_b, out);
}
